// Athena_42167988912760
// MI455X (gfx1250) — matrix operand order repaired and checked by reading, not compiled, not run on hardware
//
#include <hip/hip_runtime.h>

// ---------------------------------------------------------------------------
// Llama-style 4-layer forward pass for MI455X (gfx1250, wave32, WMMA).
// All GEMMs (incl. attention QK^T / PV) use v_wmma_f32_16x16x32_bf16.
// Weights are f32 in HBM (read once, converted to bf16 while staging to LDS);
// activations are kept in bf16 and staged HBM->LDS with
// global_load_async_to_lds_b128 (ASYNCcnt path). Workload is HBM-bound
// (~400MB weights @ 23.3 TB/s => ~17us floor).
// ---------------------------------------------------------------------------

#define SEQ     2048
#define EMBED   1024
#define NHEADS  16
#define DHEAD   64
#define HIDDEN  4096
#define VOCAB   32000
#define LAYERS  4

typedef __bf16 bf16;
typedef __bf16 v16bf __attribute__((ext_vector_type(16)));
typedef float  v8f   __attribute__((ext_vector_type(8)));

__device__ __forceinline__ bf16 to_bf16(float f) {
  unsigned u = __builtin_bit_cast(unsigned, f);
  unsigned short s = (unsigned short)((u + 0x7FFFu + ((u >> 16) & 1u)) >> 16);
  return __builtin_bit_cast(bf16, s);
}

// A-fragment (16x32, 16-bit) LDS k-index for element e of a lane:
// lanes 0-15 hold K in {0-15}, lanes 16-31 hold K in {16-31}.
__device__ __forceinline__ int a_frag_k(int e, int half) {
  return half * 16 + e;
}

// Async HBM->LDS 16B copy (gfx1250 VGLOBAL async op, tracked by ASYNCcnt).
// The low 32 bits of a flat pointer to LDS are the LDS byte address.
__device__ __forceinline__ void async_copy_b128(void* lds, const void* g) {
  unsigned l = (unsigned)(uintptr_t)lds;
  unsigned long long ga = (unsigned long long)(uintptr_t)g;
  asm volatile("global_load_async_to_lds_b128 %0, %1, off"
               :: "v"(l), "v"(ga) : "memory");
}
__device__ __forceinline__ void wait_async0() {
  asm volatile("s_wait_asynccnt 0x0" ::: "memory");
}

// ---------------------------------------------------------------------------
// Embedding gather (row 0 of the table is treated as zero, per reference).
// ---------------------------------------------------------------------------
__global__ __launch_bounds__(256)
void embed_k(const int* __restrict__ tokens, const float* __restrict__ table,
             float* __restrict__ emb) {
  int idx = blockIdx.x * 256 + threadIdx.x;       // SEQ*EMBED threads
  int s = idx >> 10, c = idx & (EMBED - 1);
  int tok = tokens[s];
  emb[idx] = (tok == 0) ? 0.0f : table[(size_t)tok * EMBED + c];
}

// ---------------------------------------------------------------------------
// RMSNorm: f32 row -> bf16 row (activation input for the next GEMM).
// ---------------------------------------------------------------------------
__global__ __launch_bounds__(256)
void rmsnorm_bf16(const float* __restrict__ X, bf16* __restrict__ Y, int ncols) {
  __shared__ float red[256];
  const int row = blockIdx.x, tid = threadIdx.x;
  const float* x = X + (size_t)row * ncols;
  float ss = 0.0f;
  for (int c = tid; c < ncols; c += 256) { float v = x[c]; ss += v * v; }
  red[tid] = ss;
  __syncthreads();
  for (int s = 128; s > 0; s >>= 1) {
    if (tid < s) red[tid] += red[tid + s];
    __syncthreads();
  }
  float scale = rsqrtf(red[0] / (float)ncols + 1e-5f);
  bf16* y = Y + (size_t)row * ncols;
  for (int c = tid; c < ncols; c += 256) y[c] = to_bf16(x[c] * scale);
}

// ---------------------------------------------------------------------------
// GEMM: C[M,N](f32) (+)= A[M,K](bf16, row-major) * B[K,N](f32 weights).
// 256 threads = 8 waves; block tile 64x64; K-step 32.
//  - A tile staged with global_load_async_to_lds_b128 (bf16, no conversion).
//  - B tile converted f32->bf16 and stored TRANSPOSED (sBt[n][k]) so every
//    B fragment is a contiguous 32B run => 2x ds_load_b128 (tile is staged
//    once but read by 8 waves, so scattered stores < scattered loads).
// ---------------------------------------------------------------------------
template <int ACCUM>
__global__ __launch_bounds__(256)
void gemm_xw(const bf16* __restrict__ A, const float* __restrict__ B,
             float* __restrict__ C, int M, int N, int K) {
  __shared__ alignas(16) bf16 sA[64][40];    // 64(M) x 32(K) tile, padded
  __shared__ alignas(16) bf16 sBt[64][40];   // 64(N) x 32(K) tile, padded
  const int tid  = threadIdx.x;
  const int lane = tid & 31, wave = tid >> 5;
  const int bm = blockIdx.y * 64, bn = blockIdx.x * 64;
  const int m = lane & 15, half = lane >> 4;
  const int mi = wave >> 1, ni0 = (wave & 1) * 2;

  v8f acc0 = {}, acc1 = {};

  for (int k0 = 0; k0 < K; k0 += 32) {
    // Stage A (bf16, 64x32): 8 elements (16B) per thread, async to LDS.
    {
      int e = tid * 8, r = e >> 5, c = e & 31;
      async_copy_b128(&sA[r][c], &A[(size_t)(bm + r) * K + k0 + c]);
    }
    // Stage B (f32 -> bf16, 32x64, transposed): 8 floats per thread.
    {
      int e = tid * 8, kk = e >> 6, c = e & 63;  // kk in [0,32), c in [0,64)
      const float4* src =
          reinterpret_cast<const float4*>(&B[(size_t)(k0 + kk) * N + bn + c]);
      float4 f0 = src[0], f1 = src[1];
      sBt[c + 0][kk] = to_bf16(f0.x); sBt[c + 1][kk] = to_bf16(f0.y);
      sBt[c + 2][kk] = to_bf16(f0.z); sBt[c + 3][kk] = to_bf16(f0.w);
      sBt[c + 4][kk] = to_bf16(f1.x); sBt[c + 5][kk] = to_bf16(f1.y);
      sBt[c + 6][kk] = to_bf16(f1.z); sBt[c + 7][kk] = to_bf16(f1.w);
      if (k0 + 32 < K)  // prefetch next weight tile (global_prefetch_b8)
        __builtin_prefetch(&B[(size_t)(k0 + 32 + kk) * N + bn + c], 0, 1);
    }
    wait_async0();
    __syncthreads();

    v16bf af, bf0, bf1;
#pragma unroll
    for (int e = 0; e < 16; ++e) {
      af[e]  = sA[mi * 16 + m][a_frag_k(e, half)];
      bf0[e] = sBt[ni0 * 16 + m][half * 16 + e];        // contiguous 32B
      bf1[e] = sBt[(ni0 + 1) * 16 + m][half * 16 + e];  // contiguous 32B
    }
    acc0 = __builtin_amdgcn_wmma_f32_16x16x32_bf16(false, af, false, bf0,
                                                   (short)0, acc0, false, false);
    acc1 = __builtin_amdgcn_wmma_f32_16x16x32_bf16(false, af, false, bf1,
                                                   (short)0, acc1, false, false);
    __syncthreads();
  }

  // Epilogue: D layout => row = 16*mi + r + 8*half, col = 16*ni + (lane&15).
#pragma unroll
  for (int r = 0; r < 8; ++r) {
    int row = bm + mi * 16 + half * 8 + r;
    size_t i0 = (size_t)row * N + bn + ni0 * 16 + m;
    size_t i1 = (size_t)row * N + bn + (ni0 + 1) * 16 + m;
    if (ACCUM) { C[i0] += acc0[r]; C[i1] += acc1[r]; }
    else       { C[i0]  = acc0[r]; C[i1]  = acc1[r]; }
  }
}

// ---------------------------------------------------------------------------
// Split qkv, apply RoPE to q/k, fold 1/sqrt(64) into q, emit bf16 head-major
// [head][seq][dhead] buffers for the attention kernel.
// ---------------------------------------------------------------------------
__global__ __launch_bounds__(256)
void rope_split(const float* __restrict__ qkv, bf16* __restrict__ Q,
                bf16* __restrict__ Kh, bf16* __restrict__ V) {
  int idx = blockIdx.x * 256 + threadIdx.x;       // SEQ*NHEADS*DHEAD threads
  int i = idx & 63, h = (idx >> 6) & 15, s = idx >> 10;
  const float* base = qkv + (size_t)s * (3 * EMBED) + h * DHEAD;
  int off = i & 31;
  float scale = __expf(-(9.210340372f / 32.0f) * (float)off); // 10000^(-off/32)
  float angle = (float)s * scale;
  float cs = __cosf(angle), sn = __sinf(angle);
  float sgn = (i < 32) ? -sn : sn;
  int ipair = (i + 32) & 63;
  float q = cs * base[i] + sgn * base[ipair];
  float k = cs * base[EMBED + i] + sgn * base[EMBED + ipair];
  float v = base[2 * EMBED + i];
  size_t o = ((size_t)h * SEQ + s) * DHEAD + i;
  Q[o]  = to_bf16(q * 0.125f);   // 1/sqrt(DHEAD) folded into Q
  Kh[o] = to_bf16(k);
  V[o]  = to_bf16(v);
}

// ---------------------------------------------------------------------------
// Flash attention: grid (SEQ/64, NHEADS); 128 threads = 4 waves; each wave
// owns 16 query rows. K/V staged to LDS per 32-key block via async copies;
// scores and PV via bf16 WMMA; online softmax with wave32 shfl reductions
// over the D-layout halves; P relayout D->A via per-wave LDS scratch
// (s_wait_dscnt fence, wave-local).
// ---------------------------------------------------------------------------
__global__ __launch_bounds__(128)
void attention_k(const bf16* __restrict__ Q, const bf16* __restrict__ Kh,
                 const bf16* __restrict__ V, bf16* __restrict__ O) {
  __shared__ alignas(16) bf16 sK[32][72];
  __shared__ alignas(16) bf16 sV[32][72];
  __shared__ bf16 sP[4][16][34];
  const int tid = threadIdx.x, lane = tid & 31, wave = tid >> 5;
  const int qb = blockIdx.x, h = blockIdx.y;
  const int m = lane & 15, half = lane >> 4;
  const int q0 = qb * 64 + wave * 16;

  // Q fragments for the two 32-wide d-chunks (loaded once; each is two
  // contiguous 16B runs per lane => global_load_b128 pairs).
  v16bf qa0, qa1;
  const bf16* qrow = Q + ((size_t)h * SEQ + q0 + m) * DHEAD;
#pragma unroll
  for (int e = 0; e < 16; ++e) {
    int k = a_frag_k(e, half);
    qa0[e] = qrow[k];
    qa1[e] = qrow[32 + k];
  }

  float mrow[8], lrow[8];
  v8f oacc[4] = {{}, {}, {}, {}};
#pragma unroll
  for (int r = 0; r < 8; ++r) { mrow[r] = -1e30f; lrow[r] = 0.0f; }

  const int jmax = 2 * qb + 2;  // key blocks of 32, through the diagonal
  for (int jb = 0; jb < jmax; ++jb) {
    const int kb0 = jb * 32;
    {   // async K/V tile staging: 16 bf16 (2x16B) per thread per matrix
      int row = tid >> 2, c = (tid & 3) * 16;
      size_t g = ((size_t)h * SEQ + kb0 + row) * DHEAD + c;
      async_copy_b128(&sK[row][c],     &Kh[g]);
      async_copy_b128(&sK[row][c + 8], &Kh[g + 8]);
      async_copy_b128(&sV[row][c],     &V[g]);
      async_copy_b128(&sV[row][c + 8], &V[g + 8]);
    }
    wait_async0();
    __syncthreads();

    // S = Q * K^T over 32 keys: two 16-key n-tiles, two 32-wide d-chunks.
    v16bf b00, b01, b10, b11;
#pragma unroll
    for (int e = 0; e < 16; ++e) {
      int d = half * 16 + e;
      b00[e] = sK[m][d];            b01[e] = sK[m][32 + d];
      b10[e] = sK[16 + m][d];       b11[e] = sK[16 + m][32 + d];
    }
    v8f s0 = {}, s1 = {};
    s0 = __builtin_amdgcn_wmma_f32_16x16x32_bf16(false, qa0, false, b00,
                                                 (short)0, s0, false, false);
    s0 = __builtin_amdgcn_wmma_f32_16x16x32_bf16(false, qa1, false, b01,
                                                 (short)0, s0, false, false);
    s1 = __builtin_amdgcn_wmma_f32_16x16x32_bf16(false, qa0, false, b10,
                                                 (short)0, s1, false, false);
    s1 = __builtin_amdgcn_wmma_f32_16x16x32_bf16(false, qa1, false, b11,
                                                 (short)0, s1, false, false);

    // Causal mask + online softmax (rows live in lane-halves of 16 lanes).
    float alpha[8];
#pragma unroll
    for (int r = 0; r < 8; ++r) {
      int qg = q0 + half * 8 + r;
      float x0 = (kb0 + m)      <= qg ? s0[r] : -1e30f;
      float x1 = (kb0 + 16 + m) <= qg ? s1[r] : -1e30f;
      float v = fmaxf(x0, x1);
#pragma unroll
      for (int d = 1; d < 16; d <<= 1) v = fmaxf(v, __shfl_xor(v, d, 32));
      float mn = fmaxf(mrow[r], v);
      alpha[r] = __expf(mrow[r] - mn);
      mrow[r] = mn;
      float p0 = __expf(x0 - mn), p1 = __expf(x1 - mn);
      sP[wave][half * 8 + r][m]      = to_bf16(p0);
      sP[wave][half * 8 + r][16 + m] = to_bf16(p1);
      float ps = p0 + p1;
#pragma unroll
      for (int d = 1; d < 16; d <<= 1) ps += __shfl_xor(ps, d, 32);
      lrow[r] = lrow[r] * alpha[r] + ps;
    }
#pragma unroll
    for (int r = 0; r < 8; ++r) {
      oacc[0][r] *= alpha[r]; oacc[1][r] *= alpha[r];
      oacc[2][r] *= alpha[r]; oacc[3][r] *= alpha[r];
    }

    // Wave-local fence: sP writes (this wave's lanes) -> sP reads below.
    asm volatile("s_wait_dscnt 0x0" ::: "memory");

    v16bf pa;
#pragma unroll
    for (int e = 0; e < 16; ++e) pa[e] = sP[wave][m][a_frag_k(e, half)];

    // O += P (16x32) * V (32x64): one WMMA per 16-wide d-tile.
#pragma unroll
    for (int dt = 0; dt < 4; ++dt) {
      v16bf vb;
#pragma unroll
      for (int e = 0; e < 16; ++e) vb[e] = sV[half * 16 + e][dt * 16 + m];
      oacc[dt] = __builtin_amdgcn_wmma_f32_16x16x32_bf16(
          false, pa, false, vb, (short)0, oacc[dt], false, false);
    }
    __syncthreads();
  }

  // Normalize and write bf16 [seq][h*64+d] for the out-projection GEMM.
#pragma unroll
  for (int r = 0; r < 8; ++r) {
    float inv = 1.0f / lrow[r];
    int row = q0 + half * 8 + r;
    bf16* orow = O + (size_t)row * (NHEADS * DHEAD) + h * DHEAD;
#pragma unroll
    for (int dt = 0; dt < 4; ++dt)
      orow[dt * 16 + m] = to_bf16(oacc[dt][r] * inv);
  }
}

// ---------------------------------------------------------------------------
// SwiGLU: h = up * silu(gate), bf16 output for the down GEMM.
// ---------------------------------------------------------------------------
__global__ __launch_bounds__(256)
void silu_mul(const float* __restrict__ gu, bf16* __restrict__ Hh) {
  int idx = blockIdx.x * 256 + threadIdx.x;       // SEQ*HIDDEN threads
  int s = idx >> 12, j = idx & (HIDDEN - 1);
  float g = gu[(size_t)s * (2 * HIDDEN) + j];
  float u = gu[(size_t)s * (2 * HIDDEN) + HIDDEN + j];
  Hh[idx] = to_bf16(u * g / (1.0f + __expf(-g)));
}

// ---------------------------------------------------------------------------
// Launch: full forward pass.
// ---------------------------------------------------------------------------
extern "C" void kernel_launch(void* const* d_in, const int* in_sizes, int n_in,
                              void* d_out, int out_size, void* d_ws,
                              size_t ws_size, hipStream_t stream) {
  (void)in_sizes; (void)n_in; (void)out_size; (void)ws_size;
  const int*   tokens  = (const int*)d_in[0];
  const float* table   = (const float*)d_in[1];
  const float* qkv_w   = (const float*)d_in[2];
  const float* out_w   = (const float*)d_in[3];
  const float* up_w    = (const float*)d_in[4];
  const float* down_w  = (const float*)d_in[5];
  const float* vocab_w = (const float*)d_in[6];
  float* out = (float*)d_out;

  char* ws = (char*)d_ws;
  size_t off = 0;
  auto alloc = [&](size_t bytes) -> void* {
    void* p = ws + off;
    off = (off + bytes + 255) & ~(size_t)255;
    return p;
  };
  float* emb    = (float*)alloc((size_t)SEQ * EMBED * 4);
  bf16*  xb     = (bf16*) alloc((size_t)SEQ * EMBED * 2);
  float* qkv    = (float*)alloc((size_t)SEQ * 3 * EMBED * 4);
  bf16*  qh     = (bf16*) alloc((size_t)SEQ * EMBED * 2);
  bf16*  kh     = (bf16*) alloc((size_t)SEQ * EMBED * 2);
  bf16*  vh     = (bf16*) alloc((size_t)SEQ * EMBED * 2);
  bf16*  ao     = (bf16*) alloc((size_t)SEQ * EMBED * 2);
  float* gateup = (float*)alloc((size_t)SEQ * 2 * HIDDEN * 4);
  bf16*  hbuf   = (bf16*) alloc((size_t)SEQ * HIDDEN * 2);

  embed_k<<<SEQ * EMBED / 256, 256, 0, stream>>>(tokens, table, emb);

  for (int l = 0; l < LAYERS; ++l) {
    rmsnorm_bf16<<<SEQ, 256, 0, stream>>>(emb, xb, EMBED);
    gemm_xw<0><<<dim3(3 * EMBED / 64, SEQ / 64), 256, 0, stream>>>(
        xb, qkv_w + (size_t)l * EMBED * 3 * EMBED, qkv, SEQ, 3 * EMBED, EMBED);
    rope_split<<<SEQ * EMBED / 256, 256, 0, stream>>>(qkv, qh, kh, vh);
    attention_k<<<dim3(SEQ / 64, NHEADS), 128, 0, stream>>>(qh, kh, vh, ao);
    gemm_xw<1><<<dim3(EMBED / 64, SEQ / 64), 256, 0, stream>>>(
        ao, out_w + (size_t)l * EMBED * EMBED, emb, SEQ, EMBED, EMBED);
    rmsnorm_bf16<<<SEQ, 256, 0, stream>>>(emb, xb, EMBED);
    gemm_xw<0><<<dim3(2 * HIDDEN / 64, SEQ / 64), 256, 0, stream>>>(
        xb, up_w + (size_t)l * EMBED * 2 * HIDDEN, gateup, SEQ, 2 * HIDDEN,
        EMBED);
    silu_mul<<<SEQ * HIDDEN / 256, 256, 0, stream>>>(gateup, hbuf);
    gemm_xw<1><<<dim3(EMBED / 64, SEQ / 64), 256, 0, stream>>>(
        hbuf, down_w + (size_t)l * HIDDEN * EMBED, emb, SEQ, EMBED, HIDDEN);
  }

  rmsnorm_bf16<<<SEQ, 256, 0, stream>>>(emb, xb, EMBED);
  gemm_xw<0><<<dim3(VOCAB / 64, SEQ / 64), 256, 0, stream>>>(
      xb, vocab_w, out, SEQ, VOCAB, EMBED);
}
